// HEPOSSummarizer_75445395522268
// MI455X (gfx1250) — matrix operand order repaired and checked by reading, not compiled, not run on hardware
//
#include <hip/hip_runtime.h>

// ---------------------------------------------------------------------------
// Types for CDNA5 WMMA (wave32): 16x16x32 f16 -> f32
// ---------------------------------------------------------------------------
typedef __attribute__((ext_vector_type(16))) _Float16 v16h;
typedef __attribute__((ext_vector_type(8)))  float    v8f;
typedef __attribute__((ext_vector_type(4)))  unsigned u32x4;
typedef __attribute__((ext_vector_type(8)))  unsigned u32x8;

__device__ __forceinline__ v8f wmma_f16(v16h a, v16h b, v8f c) {
  // (neg_a, A, neg_b, B, c_mod, C, reuse_a, reuse_b)
  return __builtin_amdgcn_wmma_f32_16x16x32_f16(false, a, false, b, (short)0, c,
                                                false, false);
}

// Raw LDS byte offset of a shared-memory pointer (addrspace(3) ptrtoint),
// as required by the TDM descriptor's lds_addr field.
__device__ __forceinline__ unsigned lds_offset(const void* p) {
  return (unsigned)(unsigned long long)(const __attribute__((address_space(3))) char*)p;
}

// ---------------------------------------------------------------------------
// Tensor Data Mover: async 2D tile load (rows x cols elements) global -> LDS.
// dsize_code: 0=1B, 1=2B, 2=4B.  LDS rows are padded by 4 dwords per 32-dword
// row (pad_interval=32dw, pad_amount=4dw): a 128B tile row lands on a 144B
// pitch -> conflict-free fragment reads (4*(9m mod 16) bank pattern).
// Tracked on TENSORcnt; issue from ONE wave, s_wait_tensorcnt before barrier.
// ---------------------------------------------------------------------------
__device__ __forceinline__ void tdm_load_2d(const void* gsrc, unsigned lds_off,
                                            unsigned row_stride_elems,
                                            unsigned rows, unsigned cols,
                                            unsigned dsize_code) {
  unsigned long long ga = (unsigned long long)gsrc;
  u32x4 g0;
  g0[0] = 1u;                                       // count=1, user descriptor
  g0[1] = lds_off;                                  // lds_addr (bytes)
  g0[2] = (unsigned)ga;                             // global_addr[31:0]
  g0[3] = (unsigned)(ga >> 32) | (2u << 30);        // global_addr[56:32] | type=2
  unsigned td0 = row_stride_elems;                  // tensor_dim0 (>= tile cols)
  unsigned td1 = rows;                              // tensor_dim1
  u32x8 g1;
  g1[0] = (dsize_code << 16)  // data_size
        | (1u << 20)          // pad_enable
        | (4u << 22)          // pad_interval: every 32 dwords (128B)
        | (3u << 25);         // pad_amount: 4 dwords (16B)
  g1[1] = (td0 & 0xFFFFu) << 16;                    // tensor_dim0[15:0]
  g1[2] = (td0 >> 16) | ((td1 & 0xFFFFu) << 16);    // td0[31:16] | td1[15:0]
  g1[3] = (td1 >> 16) | (cols << 16);               // td1[31:16] | tile_dim0
  g1[4] = rows;                                     // tile_dim1 (tile_dim2=0)
  g1[5] = row_stride_elems;                         // tensor_dim0_stride[31:0]
  g1[6] = 0u;
  g1[7] = 0u;
  asm volatile("tensor_load_to_lds %0, %1" ::"s"(g0), "s"(g1) : "memory");
}

// ---------------------------------------------------------------------------
// Fragment loaders (f16 elements).
// A-matrix 16x32 (MxK): lanes 0-15 hold M=lane, K=0..15;
//                       lanes 16-31 hold K=16..31 (contiguous).
// B-matrix 32x16 (KxN) from B^T rows (W[n][k]): lane n=lane&15;
//                       lanes 0-15 K=0..15, lanes 16-31 K=16..31 (contiguous).
// ---------------------------------------------------------------------------
__device__ __forceinline__ v16h frag_a_lds_h(const _Float16* sm, int ld) {
  int lane = threadIdx.x & 31;
  const _Float16* r = sm + (lane & 15) * ld;
  int kb = (lane >> 4) * 16;
  v16h f;
#pragma unroll
  for (int j = 0; j < 16; ++j) {
    int k = kb + j;
    f[j] = r[k];
  }
  return f;
}

__device__ __forceinline__ v16h frag_b_lds_h(const _Float16* sm, int ld) {
  int lane = threadIdx.x & 31;
  const _Float16* r = sm + (lane & 15) * ld + ((lane >> 4) << 4);
  v16h f;
#pragma unroll
  for (int j = 0; j < 16; ++j) f[j] = r[j];
  return f;
}

__device__ __forceinline__ v16h frag_b_glob_h(const _Float16* p, int ld) {
  int lane = threadIdx.x & 31;
  const _Float16* r = p + (size_t)(lane & 15) * ld + ((lane >> 4) << 4);
  v16h f;
#pragma unroll
  for (int j = 0; j < 16; ++j) f[j] = r[j];
  return f;
}

__device__ __forceinline__ float sigm(float x) { return 1.0f / (1.0f + __expf(-x)); }

// ---------------------------------------------------------------------------
// f32 -> f16 bulk conversion (n multiple of 2048; vectorized b128 in/out).
// ---------------------------------------------------------------------------
__global__ __launch_bounds__(256) void cvt_f16_kernel(const float* __restrict__ in,
                                                      _Float16* __restrict__ out,
                                                      int n) {
  int i = (blockIdx.x * 256 + threadIdx.x) * 8;
  if (i + 8 <= n) {
    float4 a = *(const float4*)(in + i);
    float4 b = *(const float4*)(in + i + 4);
    _Float16* o = out + i;
    o[0] = (_Float16)a.x; o[1] = (_Float16)a.y;
    o[2] = (_Float16)a.z; o[3] = (_Float16)a.w;
    o[4] = (_Float16)b.x; o[5] = (_Float16)b.y;
    o[6] = (_Float16)b.z; o[7] = (_Float16)b.w;
  }
}

// ---------------------------------------------------------------------------
// WMMA GEMM:  C[M,N] = act(A16[M,K] @ W16[N,K]^T + bias[N] (+ resid[M,N]))
// Block: 256 threads = 8 waves (4 m-waves x 2 n-waves). Block tile 64x64.
// K-step 64 (4 wmma between barriers), TDM double-buffered f16 staging
// (wave 0 drives the DMA engine). Requires M%64==0, N%64==0, K%64==0.
// ---------------------------------------------------------------------------
#define LDH 72  // LDS leading dim in halves (64 + TDM pad of 8 halves/row)

__global__ __launch_bounds__(256) void wmma_gemm_kernel(
    const _Float16* __restrict__ A, const _Float16* __restrict__ W,
    const float* __restrict__ bias, const float* __restrict__ resid,
    float* __restrict__ C, int M, int N, int K, int relu) {
  __shared__ _Float16 sA[2][64 * LDH];
  __shared__ _Float16 sB[2][64 * LDH];

  const int tid = threadIdx.x;
  const int lane = tid & 31;
  const int w = tid >> 5;
  const int wm = w >> 1;   // 0..3
  const int wn = w & 1;    // 0..1
  const int bm = blockIdx.y * 64;
  const int bn = blockIdx.x * 64;

  const _Float16* Abase = A + (size_t)bm * K;
  const _Float16* Wbase = W + (size_t)bn * K;
  const unsigned offA[2] = {lds_offset(&sA[0][0]), lds_offset(&sA[1][0])};
  const unsigned offB[2] = {lds_offset(&sB[0][0]), lds_offset(&sB[1][0])};

  v8f acc0 = {};
  v8f acc1 = {};

  const int nk = K / 64;
  if (w == 0) {
    tdm_load_2d(Abase, offA[0], K, 64, 64, 1u);
    tdm_load_2d(Wbase, offB[0], K, 64, 64, 1u);
  }

  for (int step = 0; step < nk; ++step) {
    const int cur = step & 1;
    if (w == 0) {
      if (step + 1 < nk) {
        const int k1 = (step + 1) * 64;
        tdm_load_2d(Abase + k1, offA[cur ^ 1], K, 64, 64, 1u);
        tdm_load_2d(Wbase + k1, offB[cur ^ 1], K, 64, 64, 1u);
        __builtin_amdgcn_s_wait_tensorcnt(2);  // current buffer's pair done
      } else {
        __builtin_amdgcn_s_wait_tensorcnt(0);
      }
    }
    __syncthreads();

    const _Float16* pa = sA[cur] + (wm * 16) * LDH;
    const _Float16* pb = sB[cur] + (wn * 32) * LDH;
    v16h a0  = frag_a_lds_h(pa, LDH);            // K 0..31
    v16h a1  = frag_a_lds_h(pa + 32, LDH);       // K 32..63
    v16h b00 = frag_b_lds_h(pb, LDH);            // n-sub 0, K 0..31
    v16h b10 = frag_b_lds_h(pb + 32, LDH);       // n-sub 0, K 32..63
    v16h b01 = frag_b_lds_h(pb + 16 * LDH, LDH); // n-sub 1, K 0..31
    v16h b11 = frag_b_lds_h(pb + 16 * LDH + 32, LDH);
    acc0 = wmma_f16(a0, b00, acc0);
    acc0 = wmma_f16(a1, b10, acc0);
    acc1 = wmma_f16(a0, b01, acc1);
    acc1 = wmma_f16(a1, b11, acc1);
    __syncthreads();
  }

  // Epilogue. C/D layout: lane col n=lane&15; VGPR r holds row (lane>>4)*8 + r.
  const int col = lane & 15;
  const int rg = (lane >> 4) * 8;
  const int gm = bm + wm * 16;
  const int gn = bn + wn * 32;
#pragma unroll
  for (int r = 0; r < 8; ++r) {
    int row = gm + rg + r;
    int c0 = gn + col, c1 = gn + 16 + col;
    float v0 = acc0[r], v1 = acc1[r];
    if (bias) { v0 += bias[c0]; v1 += bias[c1]; }
    size_t o0 = (size_t)row * N + c0, o1 = (size_t)row * N + c1;
    if (resid) { v0 += resid[o0]; v1 += resid[o1]; }
    if (relu) { v0 = fmaxf(v0, 0.0f); v1 = fmaxf(v1, 0.0f); }
    C[o0] = v0;
    C[o1] = v1;
  }
}

// ---------------------------------------------------------------------------
// Embedding + sinusoidal positional encoding: out[b,t,:] = emb[tok[b,t],:] + pe(t,:)
// ---------------------------------------------------------------------------
__global__ __launch_bounds__(128) void embed_pe_kernel(
    const int* __restrict__ tok, const float* __restrict__ emb,
    float* __restrict__ out, int L) {
  const int t = blockIdx.x, b = blockIdx.y;
  const int token = tok[b * L + t];
  const float* e = emb + (size_t)token * 512;
  float* o = out + ((size_t)b * L + t) * 512;
  for (int d = threadIdx.x; d < 512; d += 128) {
    int i2 = (d >> 1) * 2;
    float ang = (float)t * __expf((float)i2 * (-9.210340371976184f / 512.0f));
    float pe = (d & 1) ? __cosf(ang) : __sinf(ang);
    o[d] = e[d] + pe;
  }
}

// ---------------------------------------------------------------------------
// Residual add + LayerNorm over D=512:  out[row] = LN(x[row] + r[row]) * g + b
// One block (128 threads) per row; safe in-place (reads before first barrier).
// ---------------------------------------------------------------------------
__global__ __launch_bounds__(128) void add_ln_kernel(
    const float* __restrict__ x, const float* __restrict__ r,
    const float* __restrict__ g, const float* __restrict__ bb,
    float* __restrict__ out) {
  __shared__ float red[128];
  const int row = blockIdx.x, tid = threadIdx.x;
  const float* xr = x + (size_t)row * 512;
  const float* rr = r + (size_t)row * 512;
  float v[4];
  float s = 0.0f;
#pragma unroll
  for (int i = 0; i < 4; ++i) { v[i] = xr[tid * 4 + i] + rr[tid * 4 + i]; s += v[i]; }
  red[tid] = s; __syncthreads();
  for (int off = 64; off; off >>= 1) { if (tid < off) red[tid] += red[tid + off]; __syncthreads(); }
  const float mean = red[0] * (1.0f / 512.0f);
  __syncthreads();
  float s2 = 0.0f;
#pragma unroll
  for (int i = 0; i < 4; ++i) { float d = v[i] - mean; s2 += d * d; }
  red[tid] = s2; __syncthreads();
  for (int off = 64; off; off >>= 1) { if (tid < off) red[tid] += red[tid + off]; __syncthreads(); }
  const float inv = rsqrtf(red[0] * (1.0f / 512.0f) + 1e-5f);
  float* o = out + (size_t)row * 512;
#pragma unroll
  for (int i = 0; i < 4; ++i) {
    int c = tid * 4 + i;
    o[c] = (v[i] - mean) * inv * g[c] + bb[c];
  }
}

// ---------------------------------------------------------------------------
// Strided multi-head attention. One wave per (b, h, t) query row.
// Head h attends keys/values at seq indices 0, s, 2s, ... (s = STRIDES[h]).
// q/k/v/o are (B, L, row_stride) f32 with head h at column offset h*64, HD=64.
// ---------------------------------------------------------------------------
__global__ __launch_bounds__(32) void strided_attn_kernel(
    const float* __restrict__ q, int qs_, const float* __restrict__ k, int ks_,
    const float* __restrict__ v, int vs_, float* __restrict__ o, int os_,
    int T, int S) {
  __shared__ float probs[2048];
  __shared__ float qsm[64];
  const int t = blockIdx.x, h = blockIdx.y, b = blockIdx.z;
  const int STR[8] = {1, 2, 2, 4, 4, 8, 8, 16};
  const int stride = STR[h];
  const int Ls = S / stride;
  const int lane = threadIdx.x;

  const float* qrow = q + ((size_t)b * T + t) * qs_ + h * 64;
  qsm[lane] = qrow[lane];
  qsm[lane + 32] = qrow[lane + 32];
  __syncthreads();

  float mx = -1e30f;
  for (int si = lane; si < Ls; si += 32) {
    const float* kr = k + ((size_t)b * S + (size_t)si * stride) * ks_ + h * 64;
    float acc = 0.0f;
#pragma unroll
    for (int d = 0; d < 64; d += 4) {
      float4 kk = *(const float4*)(kr + d);
      acc += qsm[d] * kk.x + qsm[d + 1] * kk.y + qsm[d + 2] * kk.z + qsm[d + 3] * kk.w;
    }
    acc *= 0.125f;  // 1/sqrt(64)
    probs[si] = acc;
    mx = fmaxf(mx, acc);
  }
  for (int off = 16; off; off >>= 1) mx = fmaxf(mx, __shfl_xor(mx, off));
  __syncthreads();

  float sum = 0.0f;
  for (int si = lane; si < Ls; si += 32) {
    float e = __expf(probs[si] - mx);
    probs[si] = e;
    sum += e;
  }
  for (int off = 16; off; off >>= 1) sum += __shfl_xor(sum, off);
  const float inv = 1.0f / sum;
  __syncthreads();

  float o0 = 0.0f, o1 = 0.0f;
  for (int si = 0; si < Ls; ++si) {
    const float* vr = v + ((size_t)b * S + (size_t)si * stride) * vs_ + h * 64;
    float p = probs[si];
    o0 += p * vr[lane];
    o1 += p * vr[lane + 32];
  }
  float* orow = o + ((size_t)b * T + t) * os_ + h * 64;
  orow[lane] = o0 * inv;
  orow[lane + 32] = o1 * inv;
}

// ---------------------------------------------------------------------------
// LSTM recurrence (latency-bound): single persistent workgroup, 16 waves.
// Per step: gates(16x2048) = hF(16x512,f16 in LDS) @ Whh16^T via WMMA
// (rows 4..15 of hF are zero padding; only B=4 rows are real).
// G holds precomputed x@Wih^T + bih, layout (B, T, 2048). Hout is (B, T, 512).
// ---------------------------------------------------------------------------
__global__ __launch_bounds__(512) void lstm_seq_kernel(
    const float* __restrict__ G, const _Float16* __restrict__ Whh,
    const float* __restrict__ bhh, float* __restrict__ Hout, int T) {
  __shared__ _Float16 hF[16 * 512];  // padded h, f16
  __shared__ float cS[4 * 512];      // cell state
  __shared__ float gS[4 * 2048];     // recurrent gate contributions
  const int tid = threadIdx.x;
  const int lane = tid & 31;
  const int w = tid >> 5;  // 0..15

  for (int i = tid; i < 16 * 512; i += 512) hF[i] = (_Float16)0.0f;
  for (int i = tid; i < 4 * 512; i += 512) cS[i] = 0.0f;
  __syncthreads();

  for (int t = 0; t < T; ++t) {
    // Each wave: 8 N-tiles of 16 -> 128 tiles cover N=2048; K=512 in 16 steps.
#pragma unroll 1
    for (int nt = 0; nt < 8; ++nt) {
      const int n0 = (w * 8 + nt) * 16;
      // Hint-prefetch next tile's weights into cache (global_prefetch_b8).
      if (nt + 1 < 8)
        __builtin_prefetch(Whh + (size_t)(n0 + 16) * 512 + (lane << 4), 0, 1);
      v8f acc = {};
#pragma unroll 1
      for (int k0 = 0; k0 < 512; k0 += 32) {
        v16h a = frag_a_lds_h(hF + k0, 512);
        v16h b = frag_b_glob_h(Whh + (size_t)n0 * 512 + k0, 512);
        acc = wmma_f16(a, b, acc);
      }
      if (lane < 16) {  // rows 0..3 live in VGPRs 0..3 of lanes 0..15
#pragma unroll
        for (int r = 0; r < 4; ++r) gS[r * 2048 + n0 + lane] = acc[r];
      }
    }
    __syncthreads();

    // Activations + state update: 4*512 cells, 4 per thread.
    for (int idx = tid; idx < 2048; idx += 512) {
      const int b_ = idx >> 9, d = idx & 511;
      const float* Gb = G + ((size_t)b_ * T + t) * 2048;
      float gi = gS[b_ * 2048 + d]        + Gb[d]        + bhh[d];
      float gf = gS[b_ * 2048 + 512 + d]  + Gb[512 + d]  + bhh[512 + d];
      float gg = gS[b_ * 2048 + 1024 + d] + Gb[1024 + d] + bhh[1024 + d];
      float go = gS[b_ * 2048 + 1536 + d] + Gb[1536 + d] + bhh[1536 + d];
      float c = sigm(gf) * cS[idx] + sigm(gi) * tanhf(gg);
      float h = sigm(go) * tanhf(c);
      cS[idx] = c;
      hF[b_ * 512 + d] = (_Float16)h;
      Hout[((size_t)b_ * T + t) * 512 + d] = h;
    }
    __syncthreads();
  }
}

// ---------------------------------------------------------------------------
// Host orchestration
// ---------------------------------------------------------------------------
static inline void launch_gemm(const float* A, const float* W, const float* bias,
                               const float* resid, float* C, int M, int N, int K,
                               int relu, _Float16* A16, _Float16* W16,
                               hipStream_t s) {
  cvt_f16_kernel<<<(M * K) / 2048, 256, 0, s>>>(A, A16, M * K);
  cvt_f16_kernel<<<(N * K) / 2048, 256, 0, s>>>(W, W16, N * K);
  dim3 g(N / 64, M / 64);
  wmma_gemm_kernel<<<g, 256, 0, s>>>(A16, W16, bias, resid, C, M, N, K, relu);
}

extern "C" void kernel_launch(void* const* d_in, const int* in_sizes, int n_in,
                              void* d_out, int out_size, void* d_ws, size_t ws_size,
                              hipStream_t stream) {
  (void)in_sizes; (void)n_in; (void)out_size; (void)ws_size;
  constexpr int S = 2048, T = 512, B = 4;
  constexpr int MS = B * S;  // 8192 encoder rows
  constexpr int MT = B * T;  // 2048 decoder rows

  const int*   src    = (const int*)d_in[0];
  const int*   tgt    = (const int*)d_in[1];
  const float* emb    = (const float*)d_in[2];
  const float* qkv_w  = (const float*)d_in[3];   // (2, 1536, 512)
  const float* qkv_b  = (const float*)d_in[4];   // (2, 1536)
  const float* ao_w   = (const float*)d_in[5];   // (2, 512, 512)
  const float* ao_b   = (const float*)d_in[6];
  const float* ln_g   = (const float*)d_in[7];
  const float* ln_b   = (const float*)d_in[8];
  const float* f1_w   = (const float*)d_in[9];   // (2, 2048, 512)
  const float* f1_b   = (const float*)d_in[10];
  const float* f2_w   = (const float*)d_in[11];  // (2, 512, 2048)
  const float* f2_b   = (const float*)d_in[12];
  const float* wih    = (const float*)d_in[13];  // (2, 2048, 512)
  const float* whh    = (const float*)d_in[14];
  const float* bih    = (const float*)d_in[15];
  const float* bhh    = (const float*)d_in[16];
  const float* ca_qw  = (const float*)d_in[17];
  const float* ca_qb  = (const float*)d_in[18];
  const float* ca_kw  = (const float*)d_in[19];
  const float* ca_kb  = (const float*)d_in[20];
  const float* ca_vw  = (const float*)d_in[21];
  const float* ca_vb  = (const float*)d_in[22];
  const float* ca_ow  = (const float*)d_in[23];
  const float* ca_ob  = (const float*)d_in[24];
  const float* out_w  = (const float*)d_in[25];  // (32000, 512)
  const float* out_b  = (const float*)d_in[26];
  float* logits = (float*)d_out;

  // Workspace layout (float units). Decoder buffers alias dead encoder temps.
  float* ws   = (float*)d_ws;
  float* x    = ws;                       // 8192*512   (enc stream, live throughout)
  float* qkvb = x + 4194304;              // 8192*1536
  float* attn = qkvb + 12582912;          // 8192*512
  float* proj = attn + 4194304;           // 8192*512
  float* ffn1 = proj + 4194304;           // 8192*2048
  // decoder aliases
  float* caK  = qkvb;                     // 8192*512
  float* caV  = qkvb + 4194304;           // 8192*512
  float* G    = qkvb + 8388608;           // 2048*2048
  float* y    = ffn1;                     // 2048*512 (dec stream)
  float* h1   = ffn1 + 1048576;           // 2048*512
  float* caq  = ffn1 + 2097152;           // 2048*512
  float* ctxa = ffn1 + 3145728;           // 2048*512
  float* fin  = ffn1 + 4194304;           // 2048*512
  // f16 staging arenas
  _Float16* A16 = (_Float16*)(ws + 41943040);            // up to 8192*2048 halves
  _Float16* W16 = (_Float16*)(ws + 41943040 + 8388608);  // up to 32000*512 halves

  // ---- Encoder ----
  embed_pe_kernel<<<dim3(S, B), 128, 0, stream>>>(src, emb, x, S);
  for (int l = 0; l < 2; ++l) {
    launch_gemm(x, qkv_w + (size_t)l * 1536 * 512, qkv_b + l * 1536, nullptr,
                qkvb, MS, 1536, 512, 0, A16, W16, stream);
    strided_attn_kernel<<<dim3(S, 8, B), 32, 0, stream>>>(
        qkvb, 1536, qkvb + 512, 1536, qkvb + 1024, 1536, attn, 512, S, S);
    launch_gemm(attn, ao_w + (size_t)l * 512 * 512, ao_b + l * 512, nullptr,
                proj, MS, 512, 512, 0, A16, W16, stream);
    add_ln_kernel<<<MS, 128, 0, stream>>>(x, proj, ln_g + l * 512, ln_b + l * 512, x);
    launch_gemm(x, f1_w + (size_t)l * 2048 * 512, f1_b + l * 2048, nullptr,
                ffn1, MS, 2048, 512, 1, A16, W16, stream);
    launch_gemm(ffn1, f2_w + (size_t)l * 512 * 2048, f2_b + l * 512, nullptr,
                proj, MS, 512, 2048, 0, A16, W16, stream);
    add_ln_kernel<<<MS, 128, 0, stream>>>(x, proj, ln_g + l * 512, ln_b + l * 512, x);
  }

  // ---- Decoder LSTM ----
  embed_pe_kernel<<<dim3(T, B), 128, 0, stream>>>(tgt, emb, y, T);
  for (int l = 0; l < 2; ++l) {
    const float* in = (l == 0) ? y : h1;
    float* out = (l == 0) ? h1 : y;
    launch_gemm(in, wih + (size_t)l * 2048 * 512, bih + l * 2048, nullptr,
                G, MT, 2048, 512, 0, A16, W16, stream);
    // Whh -> f16 (reuses W16 arena; prior gemm already consumed it).
    cvt_f16_kernel<<<(2048 * 512) / 2048, 256, 0, stream>>>(
        whh + (size_t)l * 2048 * 512, W16, 2048 * 512);
    lstm_seq_kernel<<<1, 512, 0, stream>>>(G, W16, bhh + l * 2048, out, T);
  }
  // dec = y

  // ---- Cross attention ----
  launch_gemm(y, ca_qw, ca_qb, nullptr, caq, MT, 512, 512, 0, A16, W16, stream);
  launch_gemm(x, ca_kw, ca_kb, nullptr, caK, MS, 512, 512, 0, A16, W16, stream);
  launch_gemm(x, ca_vw, ca_vb, nullptr, caV, MS, 512, 512, 0, A16, W16, stream);
  strided_attn_kernel<<<dim3(T, 8, B), 32, 0, stream>>>(
      caq, 512, caK, 512, caV, 512, ctxa, 512, T, S);
  // fin = ctx @ ca_ow^T + ca_ob + dec
  launch_gemm(ctxa, ca_ow, ca_ob, y, fin, MT, 512, 512, 0, A16, W16, stream);

  // ---- Vocab projection ----
  launch_gemm(fin, out_w, out_b, nullptr, logits, MT, 32000, 512, 0, A16, W16, stream);
}
